// BiLSTMTagger_37323265802560
// MI455X (gfx1250) — matrix operand order repaired and checked by reading, not compiled, not run on hardware
//
#include <hip/hip_runtime.h>
#include <hip/hip_bf16.h>

// ---------------------------------------------------------------------------
// BiLSTM tagger on gfx1250: all GEMMs via v_wmma_f32_16x16x32_f16 (wave32).
// Round 3: 2x-unrolled k-loops with two named fragment buffers alternating
// roles -> removes the v_dual_mov ping-pong copies the compiler emitted for
// the round-2 double buffer, while keeping loads overlapped with WMMA batches.
// ---------------------------------------------------------------------------

typedef __attribute__((ext_vector_type(16))) _Float16 v16h;
typedef __attribute__((ext_vector_type(8)))  _Float16 v8h;
typedef __attribute__((ext_vector_type(4)))  _Float16 v4h;
typedef __attribute__((ext_vector_type(8)))  float    v8f;

#define WMMA16x16x32(a, b, c) \
  __builtin_amdgcn_wmma_f32_16x16x32_f16(false, (a), false, (b), (short)0, (c), false, false)

// A-matrix fragment, 16x32 f16 (M x K), per CDNA5 ISA layout:
//   lanes 0-15 : row M = lane,    K = k0+{0..7, 16..23}
//   lanes 16-31: row M = lane-16, K = k0+{8..15,24..31}
// -> two contiguous 16B loads per lane.
static __device__ __forceinline__ v16h load_a_frag(const _Float16* __restrict__ A,
                                                   int lda, int m0, int k0, int lane) {
  int l  = lane & 15;
  int kh = (lane >> 4) << 3;                 // 0 or 8
  const _Float16* p = A + (size_t)(m0 + l) * lda + (k0 + kh);
  v8h lo = *(const v8h*)(p);
  v8h hi = *(const v8h*)(p + 16);
  return __builtin_shufflevector(lo, hi, 0,1,2,3,4,5,6,7,8,9,10,11,12,13,14,15);
}

// B-matrix fragment, 32x16 f16 (K x N), from a PRE-TRANSPOSED weight Wt[K, ldn]:
//   col N = n0 + lane%16, K = k0+{0..7,16..23} (lanes 0-15) / k0+{8..15,24..31} (lanes 16-31).
static __device__ __forceinline__ v16h load_b_frag(const _Float16* __restrict__ Wt,
                                                   int ldn, int n0, int k0, int lane) {
  int kh = (lane >> 4) << 3;
  const _Float16* p = Wt + (size_t)(k0 + kh) * ldn + n0 + (lane & 15);
  v16h r;
#pragma unroll
  for (int s = 0; s < 8; ++s) {
    r[s]     = p[(size_t)s * ldn];
    r[s + 8] = p[(size_t)(16 + s) * ldn];
  }
  return r;
}

static __device__ __forceinline__ float sigm(float x) {
  return 1.0f / (1.0f + __expf(-x));
}

// Fragment-tile helpers for the 32x64 wave tile (2 A-frags, 4 B-frags).
static __device__ __forceinline__ void load_tile(const _Float16* __restrict__ A, int lda, int m0,
                                                 const _Float16* __restrict__ Bt, int ldn, int n0,
                                                 int k0, int lane,
                                                 v16h (&a)[2], v16h (&b)[4]) {
#pragma unroll
  for (int mi = 0; mi < 2; ++mi) a[mi] = load_a_frag(A, lda, m0 + mi * 16, k0, lane);
#pragma unroll
  for (int j = 0; j < 4; ++j)    b[j]  = load_b_frag(Bt, ldn, n0 + j * 16, k0, lane);
}

static __device__ __forceinline__ void mma_tile(v8f (&acc)[2][4],
                                                const v16h (&a)[2], const v16h (&b)[4]) {
#pragma unroll
  for (int mi = 0; mi < 2; ++mi)
#pragma unroll
    for (int j = 0; j < 4; ++j)
      acc[mi][j] = WMMA16x16x32(a[mi], b[j], acc[mi][j]);
}

// ---------------------------------------------------------------------------
// Weight convert+transpose: W[N,K] f32 -> Wt[K,Npad] f16 (zero-pad n>=N).
// ---------------------------------------------------------------------------
__global__ __launch_bounds__(256) void cvt_transpose(const float* __restrict__ W,
                                                     _Float16* __restrict__ Wt,
                                                     int N, int K, int Npad) {
  int i = blockIdx.x * blockDim.x + threadIdx.x;
  if (i >= K * Npad) return;
  int k = i / Npad;
  int n = i - k * Npad;
  Wt[i] = (n < N) ? (_Float16)W[(size_t)n * K + k] : (_Float16)0.f;
}

__global__ __launch_bounds__(256) void bias_sum(const float* __restrict__ a,
                                                const float* __restrict__ b,
                                                float* __restrict__ o, int n) {
  int i = blockIdx.x * blockDim.x + threadIdx.x;
  if (i < n) o[i] = a[i] + b[i];
}

// ---------------------------------------------------------------------------
// Embedding gather: X[t*B+b, e] = (f16) emb[x[b,t], e]
// ---------------------------------------------------------------------------
__global__ __launch_bounds__(128) void embed_gather(const int* __restrict__ x,
                                                    const float* __restrict__ emb,
                                                    _Float16* __restrict__ X,
                                                    int Bsz, int T, int E) {
  int r = blockIdx.x;                 // r = t*B + b
  int b = r & (Bsz - 1);
  int t = r >> 6;                     // Bsz == 64
  int idx = x[(size_t)b * T + t];
  const float* src = emb + (size_t)idx * E;
  _Float16*    dst = X   + (size_t)r   * E;
  int e = threadIdx.x * 4;            // 128 threads * 4 = 512 = E
  float4 v = *(const float4*)(src + e);
  v4h o = { (_Float16)v.x, (_Float16)v.y, (_Float16)v.z, (_Float16)v.w };
  *(v4h*)(dst + e) = o;
}

// ---------------------------------------------------------------------------
// Big NT GEMM: C[M,N] = A[M,K] * Bt[K,N] + bias[N]   (f16 in, f32 out)
// Block: 128 thr (4 waves), block tile 64x128, wave tile 32x64:
//   2 A-frags x 4 B-frags -> 8 WMMAs per 32-deep k-step.
// Two named stage buffers alternate via 2x k-unroll (no register copies):
// refill of a buffer issues right after the WMMA batch that consumed it and
// its latency hides behind the other buffer's WMMA batch.
// Requires K % 64 == 0 (true here: K = 512 or 1024).
// grid.x = N/128, grid.y = M/64.
// ---------------------------------------------------------------------------
__global__ __launch_bounds__(128) void gemm_nt_bias(const _Float16* __restrict__ A, int lda,
                                                    const _Float16* __restrict__ Bt, int ldn,
                                                    const float* __restrict__ bias,
                                                    float* __restrict__ C, int ldc,
                                                    int K) {
  int wave = threadIdx.x >> 5;
  int lane = threadIdx.x & 31;
  int m0 = blockIdx.y * 64 + (wave & 1) * 32;    // wave covers rows m0..m0+31
  int n0 = blockIdx.x * 128 + (wave >> 1) * 64;  // wave covers cols n0..n0+63

  v8f zero = {};
  v8f acc[2][4];
#pragma unroll
  for (int mi = 0; mi < 2; ++mi)
#pragma unroll
    for (int j = 0; j < 4; ++j) acc[mi][j] = zero;

  v16h a0[2], b0[4], a1[2], b1[4];
  load_tile(A, lda, m0, Bt, ldn, n0, 0,  lane, a0, b0);
  load_tile(A, lda, m0, Bt, ldn, n0, 32, lane, a1, b1);

  for (int k0 = 0; k0 < K - 64; k0 += 64) {
    mma_tile(acc, a0, b0);
    load_tile(A, lda, m0, Bt, ldn, n0, k0 + 64, lane, a0, b0);
    mma_tile(acc, a1, b1);
    load_tile(A, lda, m0, Bt, ldn, n0, k0 + 96, lane, a1, b1);
  }
  mma_tile(acc, a0, b0);
  mma_tile(acc, a1, b1);

  int nl = lane & 15;
#pragma unroll
  for (int mi = 0; mi < 2; ++mi) {
    int mrow = m0 + mi * 16 + ((lane >> 4) << 3);
#pragma unroll
    for (int j = 0; j < 4; ++j) {
      int n = n0 + j * 16 + nl;
      float bv = bias[n];
#pragma unroll
      for (int g = 0; g < 8; ++g)
        C[(size_t)(mrow + g) * ldc + n] = acc[mi][j][g] + bv;
    }
  }
}

// ---------------------------------------------------------------------------
// One LSTM timestep, both directions (grid.y = dir), one layer.
// gates[b,n] = gates_in[row(tt,b), n] + h_in[b,:] . Whh_t[:, n]
// Each workgroup owns a 64-wide hidden slice; wave w: rows 16*(w&3),
// hidden cols (w>>2)*32, 4 gates x 2 n-subtiles = 8 accumulators whose lane
// slots line up -> elementwise LSTM cell update in registers.
// k-loop unrolled 2x with two named A registers (no carried copies); the 8
// B-fragment loads of each half batch into one clause before its WMMA batch.
// ---------------------------------------------------------------------------
__global__ __launch_bounds__(256) void lstm_step(const float* __restrict__ gatesF,
                                                 const float* __restrict__ gatesB,
                                                 const _Float16* __restrict__ WhhFt,
                                                 const _Float16* __restrict__ WhhBt,
                                                 const _Float16* __restrict__ hinF,
                                                 _Float16* __restrict__ houtF,
                                                 const _Float16* __restrict__ hinB,
                                                 _Float16* __restrict__ houtB,
                                                 float* __restrict__ cF,
                                                 float* __restrict__ cB,
                                                 _Float16* __restrict__ hs,   // [T*B, 2H]
                                                 int t, int T) {
  const int Hd = 512, G = 2048, Bsz = 64;
  int dir = blockIdx.y;
  int tt  = dir ? (T - 1 - t) : t;
  const float*    gin  = dir ? gatesB : gatesF;
  const _Float16* Wt   = dir ? WhhBt  : WhhFt;
  const _Float16* hin  = dir ? hinB   : hinF;
  _Float16*       hout = dir ? houtB  : houtF;
  float*          cbuf = dir ? cB     : cF;

  int wave = threadIdx.x >> 5;
  int lane = threadIdx.x & 31;
  int mt = wave & 3;                       // 16-row tile within B=64
  int hh = (wave >> 2) * 32;               // 32-col half of 64-wide slice
  int h0 = blockIdx.x * 64;
  int nbase = h0 + hh;                     // first of two 16-col subtiles

  v8f zero = {};
  v8f acc[4][2];
#pragma unroll
  for (int ga = 0; ga < 4; ++ga)
#pragma unroll
    for (int nc = 0; nc < 2; ++nc) acc[ga][nc] = zero;

  v16h aEven = load_a_frag(hin, Hd, mt * 16, 0,  lane);
  v16h aOdd  = load_a_frag(hin, Hd, mt * 16, 32, lane);
  for (int k0 = 0; k0 < Hd; k0 += 64) {
    {
      v16h b[8];
#pragma unroll
      for (int ga = 0; ga < 4; ++ga)
#pragma unroll
        for (int nc = 0; nc < 2; ++nc)
          b[ga * 2 + nc] = load_b_frag(Wt, G, ga * Hd + nbase + nc * 16, k0, lane);
#pragma unroll
      for (int ga = 0; ga < 4; ++ga)
#pragma unroll
        for (int nc = 0; nc < 2; ++nc)
          acc[ga][nc] = WMMA16x16x32(aEven, b[ga * 2 + nc], acc[ga][nc]);
      if (k0 + 64 < Hd) aEven = load_a_frag(hin, Hd, mt * 16, k0 + 64, lane);
    }
    {
      v16h b[8];
#pragma unroll
      for (int ga = 0; ga < 4; ++ga)
#pragma unroll
        for (int nc = 0; nc < 2; ++nc)
          b[ga * 2 + nc] = load_b_frag(Wt, G, ga * Hd + nbase + nc * 16, k0 + 32, lane);
#pragma unroll
      for (int ga = 0; ga < 4; ++ga)
#pragma unroll
        for (int nc = 0; nc < 2; ++nc)
          acc[ga][nc] = WMMA16x16x32(aOdd, b[ga * 2 + nc], acc[ga][nc]);
      if (k0 + 96 < Hd) aOdd = load_a_frag(hin, Hd, mt * 16, k0 + 96, lane);
    }
  }

  int mrow = mt * 16 + ((lane >> 4) << 3);
  int nl   = lane & 15;
  size_t rowbase = (size_t)tt * Bsz;
#pragma unroll
  for (int nc = 0; nc < 2; ++nc) {
    int hcol = nbase + nc * 16 + nl;
#pragma unroll
    for (int g = 0; g < 8; ++g) {
      int b = mrow + g;
      size_t r = rowbase + b;
      const float* grow = gin + r * G;
      float gi = acc[0][nc][g] + grow[          hcol];
      float gf = acc[1][nc][g] + grow[    Hd  + hcol];
      float gg = acc[2][nc][g] + grow[2 * Hd  + hcol];
      float go = acc[3][nc][g] + grow[3 * Hd  + hcol];
      float cp = cbuf[(size_t)b * Hd + hcol];
      float cn = sigm(gf) * cp + sigm(gi) * tanhf(gg);
      float hn = sigm(go) * tanhf(cn);
      cbuf[(size_t)b * Hd + hcol] = cn;
      hout[(size_t)b * Hd + hcol] = (_Float16)hn;
      hs[r * (2 * Hd) + dir * Hd + hcol] = (_Float16)hn;
    }
  }
}

// ---------------------------------------------------------------------------
// FC: out[b,t,tag] = hs2[t*B+b,:1024] . fcWt[:1024, tag] + fc_b[tag]
// N padded 50->64; masked store with [T,B] -> [B,T] permute.
// ---------------------------------------------------------------------------
__global__ __launch_bounds__(128) void fc_gemm(const _Float16* __restrict__ A,   // [16384,1024]
                                               const _Float16* __restrict__ Wt,  // [1024,64]
                                               const float* __restrict__ bias,   // [50]
                                               float* __restrict__ out,          // [64,256,50]
                                               int TAGS) {
  const int K = 1024, Npad = 64, T = 256;
  int wave = threadIdx.x >> 5;
  int lane = threadIdx.x & 31;
  int m0 = blockIdx.x * 64 + wave * 16;

  v8f zero = {};
  v8f acc[4] = { zero, zero, zero, zero };
  for (int k0 = 0; k0 < K; k0 += 32) {
    v16h a = load_a_frag(A, K, m0, k0, lane);
    v16h b[4];
#pragma unroll
    for (int j = 0; j < 4; ++j) b[j] = load_b_frag(Wt, Npad, j * 16, k0, lane);
#pragma unroll
    for (int j = 0; j < 4; ++j) acc[j] = WMMA16x16x32(a, b[j], acc[j]);
  }

  int mrow = m0 + ((lane >> 4) << 3);
  int nl   = lane & 15;
#pragma unroll
  for (int j = 0; j < 4; ++j) {
    int n = j * 16 + nl;
    if (n >= TAGS) continue;
    float bv = bias[n];
#pragma unroll
    for (int g = 0; g < 8; ++g) {
      int r = mrow + g;            // r = t*64 + b
      int b = r & 63;
      int t = r >> 6;
      out[((size_t)b * T + t) * TAGS + n] = acc[j][g] + bv;
    }
  }
}

// ---------------------------------------------------------------------------
extern "C" void kernel_launch(void* const* d_in, const int* in_sizes, int n_in,
                              void* d_out, int out_size, void* d_ws, size_t ws_size,
                              hipStream_t stream) {
  (void)in_sizes; (void)n_in; (void)out_size; (void)ws_size;
  const int Bsz = 64, T = 256, E = 512, Hd = 512, G = 2048, TAGS = 50;
  const int TB = T * Bsz;  // 16384

  const int*   x      = (const int*)  d_in[0];
  const float* emb    = (const float*)d_in[2];
  const float* Wih_f1 = (const float*)d_in[3];
  const float* Whh_f1 = (const float*)d_in[4];
  const float* bih_f1 = (const float*)d_in[5];
  const float* bhh_f1 = (const float*)d_in[6];
  const float* Wih_b1 = (const float*)d_in[7];
  const float* Whh_b1 = (const float*)d_in[8];
  const float* bih_b1 = (const float*)d_in[9];
  const float* bhh_b1 = (const float*)d_in[10];
  const float* Wih_f2 = (const float*)d_in[11];
  const float* Whh_f2 = (const float*)d_in[12];
  const float* bih_f2 = (const float*)d_in[13];
  const float* bhh_f2 = (const float*)d_in[14];
  const float* Wih_b2 = (const float*)d_in[15];
  const float* Whh_b2 = (const float*)d_in[16];
  const float* bih_b2 = (const float*)d_in[17];
  const float* bhh_b2 = (const float*)d_in[18];
  const float* fc_W   = (const float*)d_in[19];
  const float* fc_b   = (const float*)d_in[20];
  float* out = (float*)d_out;

  // ---- workspace carve (256B aligned bump allocator) ----
  char* w = (char*)d_ws;
  auto alloc = [&](size_t bytes) -> void* {
    void* p = (void*)w;
    w += (bytes + 255) & ~(size_t)255;
    return p;
  };
  _Float16* X1   = (_Float16*)alloc((size_t)TB * E * 2);        // 16 MB
  _Float16* hs1  = (_Float16*)alloc((size_t)TB * 2 * Hd * 2);   // 32 MB
  _Float16* hs2  = (_Float16*)alloc((size_t)TB * 2 * Hd * 2);   // 32 MB
  _Float16* WihF1t = (_Float16*)alloc((size_t)512  * G * 2);
  _Float16* WihB1t = (_Float16*)alloc((size_t)512  * G * 2);
  _Float16* WhhF1t = (_Float16*)alloc((size_t)512  * G * 2);
  _Float16* WhhB1t = (_Float16*)alloc((size_t)512  * G * 2);
  _Float16* WihF2t = (_Float16*)alloc((size_t)1024 * G * 2);
  _Float16* WihB2t = (_Float16*)alloc((size_t)1024 * G * 2);
  _Float16* WhhF2t = (_Float16*)alloc((size_t)512  * G * 2);
  _Float16* WhhB2t = (_Float16*)alloc((size_t)512  * G * 2);
  _Float16* fcWt   = (_Float16*)alloc((size_t)1024 * 64 * 2);
  float* bsF1 = (float*)alloc(G * 4);
  float* bsB1 = (float*)alloc(G * 4);
  float* bsF2 = (float*)alloc(G * 4);
  float* bsB2 = (float*)alloc(G * 4);
  float* gatesF = (float*)alloc((size_t)TB * G * 4);            // 128 MB
  float* gatesB = (float*)alloc((size_t)TB * G * 4);            // 128 MB
  char* hc_base = w;
  _Float16* hF0 = (_Float16*)alloc((size_t)Bsz * Hd * 2);
  _Float16* hF1 = (_Float16*)alloc((size_t)Bsz * Hd * 2);
  _Float16* hB0 = (_Float16*)alloc((size_t)Bsz * Hd * 2);
  _Float16* hB1 = (_Float16*)alloc((size_t)Bsz * Hd * 2);
  float* cF = (float*)alloc((size_t)Bsz * Hd * 4);
  float* cB = (float*)alloc((size_t)Bsz * Hd * 4);
  size_t hc_bytes = (size_t)(w - hc_base);

  // ---- 1. weight convert + transpose, bias sums ----
  cvt_transpose<<<(512  * G) / 256, 256, 0, stream>>>(Wih_f1, WihF1t, G, 512,  G);
  cvt_transpose<<<(512  * G) / 256, 256, 0, stream>>>(Wih_b1, WihB1t, G, 512,  G);
  cvt_transpose<<<(512  * G) / 256, 256, 0, stream>>>(Whh_f1, WhhF1t, G, 512,  G);
  cvt_transpose<<<(512  * G) / 256, 256, 0, stream>>>(Whh_b1, WhhB1t, G, 512,  G);
  cvt_transpose<<<(1024 * G) / 256, 256, 0, stream>>>(Wih_f2, WihF2t, G, 1024, G);
  cvt_transpose<<<(1024 * G) / 256, 256, 0, stream>>>(Wih_b2, WihB2t, G, 1024, G);
  cvt_transpose<<<(512  * G) / 256, 256, 0, stream>>>(Whh_f2, WhhF2t, G, 512,  G);
  cvt_transpose<<<(512  * G) / 256, 256, 0, stream>>>(Whh_b2, WhhB2t, G, 512,  G);
  cvt_transpose<<<(1024 * 64) / 256, 256, 0, stream>>>(fc_W, fcWt, TAGS, 1024, 64);
  bias_sum<<<G / 256, 256, 0, stream>>>(bih_f1, bhh_f1, bsF1, G);
  bias_sum<<<G / 256, 256, 0, stream>>>(bih_b1, bhh_b1, bsB1, G);
  bias_sum<<<G / 256, 256, 0, stream>>>(bih_f2, bhh_f2, bsF2, G);
  bias_sum<<<G / 256, 256, 0, stream>>>(bih_b2, bhh_b2, bsB2, G);

  // ---- 2. embedding gather -> f16 X1 [T*B, E] ----
  embed_gather<<<TB, 128, 0, stream>>>(x, emb, X1, Bsz, T, E);

  // ---- 3. layer-1 input projections (both directions) ----
  gemm_nt_bias<<<dim3(G / 128, TB / 64), 128, 0, stream>>>(X1, E, WihF1t, G, bsF1, gatesF, G, E);
  gemm_nt_bias<<<dim3(G / 128, TB / 64), 128, 0, stream>>>(X1, E, WihB1t, G, bsB1, gatesB, G, E);

  // ---- 4. layer-1 recurrent scan (ping-pong h) ----
  hipMemsetAsync(hc_base, 0, hc_bytes, stream);
  for (int t = 0; t < T; ++t) {
    _Float16* hinF  = (t & 1) ? hF1 : hF0;
    _Float16* houtF = (t & 1) ? hF0 : hF1;
    _Float16* hinB  = (t & 1) ? hB1 : hB0;
    _Float16* houtB = (t & 1) ? hB0 : hB1;
    lstm_step<<<dim3(Hd / 64, 2), 256, 0, stream>>>(gatesF, gatesB, WhhF1t, WhhB1t,
                                                    hinF, houtF, hinB, houtB,
                                                    cF, cB, hs1, t, T);
  }

  // ---- 5. layer-2 input projections (K = 2H = 1024) ----
  gemm_nt_bias<<<dim3(G / 128, TB / 64), 128, 0, stream>>>(hs1, 2 * Hd, WihF2t, G, bsF2, gatesF, G, 2 * Hd);
  gemm_nt_bias<<<dim3(G / 128, TB / 64), 128, 0, stream>>>(hs1, 2 * Hd, WihB2t, G, bsB2, gatesB, G, 2 * Hd);

  // ---- 6. layer-2 recurrent scan ----
  hipMemsetAsync(hc_base, 0, hc_bytes, stream);
  for (int t = 0; t < T; ++t) {
    _Float16* hinF  = (t & 1) ? hF1 : hF0;
    _Float16* houtF = (t & 1) ? hF0 : hF1;
    _Float16* hinB  = (t & 1) ? hB1 : hB0;
    _Float16* houtB = (t & 1) ? hB0 : hB1;
    lstm_step<<<dim3(Hd / 64, 2), 256, 0, stream>>>(gatesF, gatesB, WhhF2t, WhhB2t,
                                                    hinF, houtF, hinB, houtB,
                                                    cF, cB, hs2, t, T);
  }

  // ---- 7. FC + permute to [B, T, TAGS] ----
  fc_gemm<<<TB / 64, 128, 0, stream>>>(hs2, fcWt, fc_b, out, TAGS);
}
